// CCLoss_55087250538870
// MI455X (gfx1250) — matrix operand order repaired and checked by reading, not compiled, not run on hardware
//
#include <hip/hip_runtime.h>
#include <hip/hip_bf16.h>

typedef __attribute__((ext_vector_type(16))) _Float16 v16h;
typedef __attribute__((ext_vector_type(8)))  float    v8f;

#define WMMA_F16(a, b, c) \
    __builtin_amdgcn_wmma_f32_16x16x32_f16(false, (a), false, (b), (short)0, (c), false, false)

#define HW 512
#define OUTD 510

// ---------------------------------------------------------------------------
// Pre-kernel: materialize the 3 banded B matrices (32x16 f16, WMMA B layout)
// from the 9 conv weights. B layout: lane n<16 = column n holds K={0..7,16..23};
// lanes>=16 hold K={8..15,24..31}; 16 halves per lane. Stored as [dy][lane][16] halves.
// ---------------------------------------------------------------------------
__global__ __launch_bounds__(96) void build_bmat_kernel(
    const float* __restrict__ Wc, _Float16* __restrict__ bout)
{
    const int tid  = threadIdx.x;          // 0..95
    const int dy   = tid >> 5;
    const int lane = tid & 31;
    const int col  = lane & 15;
    const int kb   = (lane < 16) ? 0 : 8;

    _Float16 w0 = (_Float16)Wc[dy * 3 + 0];
    _Float16 w1 = (_Float16)Wc[dy * 3 + 1];
    _Float16 w2 = (_Float16)Wc[dy * 3 + 2];

    v16h b;
#pragma unroll
    for (int j = 0; j < 16; ++j) {
        int d = kb + j + ((j >= 8) ? 8 : 0) - col;              // diagonal index
        _Float16 v = (_Float16)0.0f;
        v = (d == 0) ? w0 : v;
        v = (d == 1) ? w1 : v;
        v = (d == 2) ? w2 : v;
        b[j] = v;
    }
    *(v16h*)(bout + (size_t)(dy * 32 + lane) * 16) = b;
}

// ---------------------------------------------------------------------------
// Main kernel: one wave computes one 16x16 output tile of the NCC map via
// 15 WMMAs: for dy in 0..2, for each of 5 maps (I, J, I^2, J^2, I*J):
//   acc_m += A(rows y+dy, 32 cols, f16) x Bband_dy(32x16 banded weights, f16)
// Accumulators start as inline-0 SRC2; conv bias is added in the packed-f32
// epilogue.
// ---------------------------------------------------------------------------
__global__ __launch_bounds__(256) void ncc_tile_kernel(
    const float* __restrict__ img, const float* __restrict__ tgt,
    const _Float16* __restrict__ bmats, const float* __restrict__ bc,
    float* __restrict__ blockSums, int nImages)
{
    const int lane = threadIdx.x & 31;
    const int wave = threadIdx.x >> 5;
    const int tile = blockIdx.x * 8 + wave;          // nImages*32*32 tiles total
    const int imgIdx = tile >> 10;                   // 1024 tiles per image
    const int ty = (tile >> 5) & 31;
    const int tx = tile & 31;
    const int y0 = ty << 4;
    const int x0 = tx << 4;

    const float bias = bc[0];
    const int col = lane & 15;

    // Load the 3 precomputed banded B matrices (2 x b128 each, L2-hot)
    v16h bmat[3];
#pragma unroll
    for (int dy = 0; dy < 3; ++dy)
        bmat[dy] = *(const v16h*)(bmats + (size_t)(dy * 32 + lane) * 16);

    // A layout (16x32 f16): lane L holds row M=L&15; K-chunks {0..7,16..23}
    // (L<16) or {8..15,24..31} (L>=16) => two contiguous 8-float chunks.
    const size_t ibase = (size_t)imgIdx * (size_t)HW * (size_t)HW;
    const int c0 = x0 + ((lane < 16) ? 0 : 8);       // multiple of 8, <=504
    int c1 = c0 + 16;                                // may run past row at last tile
    c1 = (c1 > HW - 8) ? (HW - 8) : c1;              // clamp: only corrupts masked outputs
    const int rbase = y0 + (lane & 15);

    // Zero-init folds into the WMMA inline SRC2=0 constant.
    v8f acc[5] = {};

#pragma unroll
    for (int dy = 0; dy < 3; ++dy) {
        int row = rbase + dy;
        row = (row > HW - 1) ? (HW - 1) : row;       // clamp: only corrupts masked outputs
        const float* ip = img + ibase + (size_t)row * HW;
        const float* tp = tgt + ibase + (size_t)row * HW;

        float4 a0 = *(const float4*)(ip + c0);
        float4 a1 = *(const float4*)(ip + c0 + 4);
        float4 a2 = *(const float4*)(ip + c1);
        float4 a3 = *(const float4*)(ip + c1 + 4);
        float4 b0 = *(const float4*)(tp + c0);
        float4 b1 = *(const float4*)(tp + c0 + 4);
        float4 b2 = *(const float4*)(tp + c1);
        float4 b3 = *(const float4*)(tp + c1 + 4);

        float fi[16] = {a0.x, a0.y, a0.z, a0.w, a1.x, a1.y, a1.z, a1.w,
                        a2.x, a2.y, a2.z, a2.w, a3.x, a3.y, a3.z, a3.w};
        float ft[16] = {b0.x, b0.y, b0.z, b0.w, b1.x, b1.y, b1.z, b1.w,
                        b2.x, b2.y, b2.z, b2.w, b3.x, b3.y, b3.z, b3.w};

        // Convert only I and J to f16 (packed cvt), derive products in packed f16.
        v16h aI, aJ;
#pragma unroll
        for (int j = 0; j < 16; ++j) {
            aI[j] = (_Float16)fi[j];
            aJ[j] = (_Float16)ft[j];
        }
        v16h aI2 = aI * aI;                          // v_pk_mul_f16
        v16h aJ2 = aJ * aJ;
        v16h aIJ = aI * aJ;

        acc[0] = WMMA_F16(aI,  bmat[dy], acc[0]);
        acc[1] = WMMA_F16(aJ,  bmat[dy], acc[1]);
        acc[2] = WMMA_F16(aI2, bmat[dy], acc[2]);
        acc[3] = WMMA_F16(aJ2, bmat[dy], acc[3]);
        acc[4] = WMMA_F16(aIJ, bmat[dy], acc[4]);
    }

    // ---- epilogue: NCC per output pixel, vectorized packed-f32 -------------
    // C/D layout: lane owns column N=lane&15, rows M = e + (lane<16 ? 0 : 8).
    const float inv81 = 1.0f / 81.0f;                // reference hardcodes 9*9 window
    const float EPSf  = 2.220446049250313e-16f;

    v8f Is  = acc[0] + bias;                         // v_pk_add_f32
    v8f Js  = acc[1] + bias;
    v8f I2s = acc[2] + bias;
    v8f J2s = acc[3] + bias;
    v8f IJs = acc[4] + bias;
    v8f Isn = Is * inv81;
    v8f Jsn = Js * inv81;
    v8f cross = IJs - Isn * Js;                      // == IJs - Is*Js/81 (pk_fma)
    v8f Iv    = I2s - Isn * Is;
    v8f Jv    = J2s - Jsn * Js;
    v8f den   = Iv * Jv + EPSf;
    v8f num   = cross * cross;
    v8f cc;
#pragma unroll
    for (int e = 0; e < 8; ++e)
        cc[e] = num[e] * __builtin_amdgcn_rcpf(den[e]);

    float lsum;
    if ((x0 < 496) && (y0 < 496)) {
        // Interior tile: every output valid; branch is wave-uniform.
        lsum = ((cc[0] + cc[1]) + (cc[2] + cc[3])) +
               ((cc[4] + cc[5]) + (cc[6] + cc[7]));
    } else {
        // Edge tile: mask outputs beyond the 510x510 valid conv region.
        const int gx = x0 + col;
        const int rofs = (lane < 16) ? 0 : 8;
        lsum = 0.0f;
#pragma unroll
        for (int e = 0; e < 8; ++e) {
            int gy = y0 + rofs + e;
            bool valid = (gx < OUTD) && (gy < OUTD);
            lsum += valid ? cc[e] : 0.0f;
        }
    }

    // wave32 reduce, then deterministic per-block sum (no atomics)
#pragma unroll
    for (int off = 16; off > 0; off >>= 1)
        lsum += __shfl_xor(lsum, off, 32);

    __shared__ float sW[8];
    if (lane == 0) sW[wave] = lsum;
    __syncthreads();
    if (threadIdx.x == 0) {
        float s = 0.0f;
#pragma unroll
        for (int i = 0; i < 8; ++i) s += sW[i];
        blockSums[blockIdx.x] = s;
    }
}

// Deterministic fixed-order reduction of block partials -> final scalar.
__global__ __launch_bounds__(256) void ncc_finalize(
    const float* __restrict__ blockSums, int n, float* __restrict__ out, float scale)
{
    __shared__ float sm[256];
    float s = 0.0f;
    for (int i = threadIdx.x; i < n; i += 256) s += blockSums[i];  // fixed stride order
    sm[threadIdx.x] = s;
    __syncthreads();
    for (int st = 128; st > 0; st >>= 1) {
        if ((int)threadIdx.x < st) sm[threadIdx.x] += sm[threadIdx.x + st];
        __syncthreads();
    }
    if (threadIdx.x == 0) out[0] = -sm[0] * scale;
}

extern "C" void kernel_launch(void* const* d_in, const int* in_sizes, int n_in,
                              void* d_out, int out_size, void* d_ws, size_t ws_size,
                              hipStream_t stream) {
    const float* img = (const float*)d_in[0];   // [B,1,512,512] f32
    const float* tgt = (const float*)d_in[1];   // [B,1,512,512] f32
    const float* Wc  = (const float*)d_in[2];   // [1,1,3,3] f32
    const float* bc  = (const float*)d_in[3];   // [1] f32
    float* out = (float*)d_out;

    const int nImages = in_sizes[0] / (HW * HW);        // 32
    const int tiles   = nImages * 32 * 32;              // 16x16 tiles over 512x512
    const int blocks  = tiles / 8;                      // 8 waves (tiles) per block

    float*     bsum  = (float*)d_ws;                    // blocks * 4 bytes
    _Float16*  bmats = (_Float16*)((char*)d_ws + 65536); // 3*32*16 halves = 3 KB

    build_bmat_kernel<<<1, 96, 0, stream>>>(Wc, bmats);
    ncc_tile_kernel<<<blocks, 256, 0, stream>>>(img, tgt, bmats, bc, bsum, nImages);

    const float scale = 1.0f / ((float)nImages * (float)OUTD * (float)OUTD);
    ncc_finalize<<<1, 256, 0, stream>>>(bsum, blocks, out, scale);
}
